// PyGTypeSpecificGAT_80075370266775
// MI455X (gfx1250) — compile-verified, matrix operand order checked by reading, not run on hardware
//
#include <hip/hip_runtime.h>
#include <hip/hip_bf16.h>
#include <math.h>

// ---------------------------------------------------------------------------
// CDNA5 (gfx1250) relational GAT: bf16 WMMA GEMMs with TDM (tensor_load_to_lds)
// B-tile staging and double-buffered LDS pipeline + L2-resident edge passes.
// MI455X: 192MB L2 holds X/h/acc (51MB each), so edge gathers/atomics hit L2.
// ---------------------------------------------------------------------------

typedef __attribute__((ext_vector_type(16))) __bf16 bf16x16;
typedef __attribute__((ext_vector_type(8)))  __bf16 bf16x8;
typedef __attribute__((ext_vector_type(4)))  __bf16 bf16x4;
typedef __attribute__((ext_vector_type(8)))  float  f32x8;
typedef __attribute__((ext_vector_type(4)))  unsigned int u32x4;
typedef __attribute__((ext_vector_type(8)))  int i32x8;
typedef __attribute__((ext_vector_type(4)))  int i32x4;

#if defined(__has_builtin)
#  if __has_builtin(__builtin_amdgcn_tensor_load_to_lds)
#    define USE_TDM 1
#  endif
#endif
#ifndef USE_TDM
#  define USE_TDM 0
#endif

// Monotonic float<->uint mapping so atomicMax(u32) implements float max
// (leaky_relu outputs can be negative).
__device__ __forceinline__ unsigned int fenc(float f) {
  unsigned int u = __float_as_uint(f);
  return (u & 0x80000000u) ? ~u : (u | 0x80000000u);
}
__device__ __forceinline__ float fdec(unsigned int u) {
  u = (u & 0x80000000u) ? (u & 0x7FFFFFFFu) : ~u;
  return __uint_as_float(u);
}
#define FENC_NEG_INF 0x007FFFFFu   // fenc(-inf)

// ---------------------------------------------------------------------------
// One-shot: W [256 k][256 n] f32  ->  Wt [256 n][256 k] bf16.
// ---------------------------------------------------------------------------
__global__ __launch_bounds__(256)
void conv_transpose_w(const float* __restrict__ W, __bf16* __restrict__ Wt) {
  __shared__ float tile[32][33];
  const int kb = blockIdx.x * 32, nb = blockIdx.y * 32;
  const int tx = threadIdx.x & 31, ty = threadIdx.x >> 5;
#pragma unroll
  for (int i = 0; i < 32; i += 8)
    tile[ty + i][tx] = W[(size_t)(kb + ty + i) * 256 + nb + tx];
  __syncthreads();
#pragma unroll
  for (int i = 0; i < 32; i += 8)
    Wt[(size_t)(nb + ty + i) * 256 + kb + tx] = (__bf16)tile[tx][ty + i];
}

// ---------------------------------------------------------------------------
// C[M,256] = A[M,256] * B, A f32 (converted on the fly), Bt bf16 [n][k].
// Block = 256 threads (8 waves), 128x128 tile, K-chunk 32, double-buffered.
// Wave (wm,wn) in 4x2 grid owns a 32x64 sub-tile = 2x4 wmma accumulators.
// B tile (8KB bf16) staged per K-chunk by the Tensor Data Mover.
// ---------------------------------------------------------------------------
__global__ __launch_bounds__(256)
void wmma_gemm_256(const float* __restrict__ A, const __bf16* __restrict__ Bt,
                   float* __restrict__ C, const float* __restrict__ zpad, int M) {
  __shared__ __bf16 lA[2][128][32];   // [buf][m][k]  2 x 8KB
  __shared__ __bf16 lB[2][128][32];   // [buf][n][k]  2 x 8KB

  const int m0   = blockIdx.x * 128;
  const int n0   = blockIdx.y * 128;
  const int tid  = threadIdx.x;
  const int wave = tid >> 5;
  const int lane = tid & 31;
  const int wm   = wave & 3;          // 0..3 along M
  const int wn   = wave >> 2;         // 0..1 along N
  const int l15  = lane & 15;
  const int rsel = lane >> 4;         // K base 0/8 per ISA 16-bit frag layout

  f32x8 acc[2][4];
#pragma unroll
  for (int i = 0; i < 2; ++i)
#pragma unroll
    for (int j = 0; j < 4; ++j) acc[i][j] = f32x8{};

  // ---- A staging: row bound is loop-invariant -> precomputed pointers,
  //      OOB rows read a zero page (no per-chunk cndmask). ----
  const float* ap[4];
#pragma unroll
  for (int p = 0; p < 4; ++p) {
    int f = tid + p * 256;            // float4 id, 8 per row
    int r = f >> 3, c4 = f & 7;
    int gr = m0 + r;
    ap[p] = (gr < M ? A + (size_t)gr * 256 : zpad) + c4 * 4;
  }

  float4 ra[4];
  auto gloadA = [&](int k0) {
#pragma unroll
    for (int p = 0; p < 4; ++p) ra[p] = *(const float4*)(ap[p] + k0);
  };
  auto storeA = [&](int buf) {
#pragma unroll
    for (int p = 0; p < 4; ++p) {
      int f = tid + p * 256;
      int r = f >> 3, c4 = f & 7;
      bf16x4 bv;
      bv[0] = (__bf16)ra[p].x; bv[1] = (__bf16)ra[p].y;
      bv[2] = (__bf16)ra[p].z; bv[3] = (__bf16)ra[p].w;
      *(bf16x4*)(&lA[buf][r][c4 * 4]) = bv;        // ds_store_b64
    }
  };

#if USE_TDM
  // ---- B staging via Tensor Data Mover: 2D tile [128 n][32 k] bf16 from
  //      Wt (tensor 256x256, dim0=k contiguous, stride 256) into lB[buf].
  //      Tile rows land contiguously in LDS = exactly lB[buf]'s layout. ----
  auto tdmB = [&](int buf, int k0) {
    unsigned lds = (unsigned)(uintptr_t)(void*)&lB[buf][0][0];  // low 32b = LDS offset
    unsigned long long ga =
        (unsigned long long)(uintptr_t)(Bt + (size_t)n0 * 256 + k0);
    u32x4 g0;
    g0[0] = 1u;                                  // count=1, user descriptor
    g0[1] = lds;                                 // lds_addr
    g0[2] = (unsigned)ga;                        // global_addr[31:0]
    g0[3] = (unsigned)((ga >> 32) & 0x01FFFFFFull) | 0x80000000u; // [56:32]+type=2
    i32x8 g1;
    g1[0] = 0x00010000;                          // mask=0, data_size=1 (2B)
    g1[1] = (int)(256u << 16);                   // tensor_dim0[15:0]=256
    g1[2] = (int)(256u << 16);                   // td0 hi=0 | tensor_dim1[15:0]=256
    g1[3] = (int)(32u << 16);                    // td1 hi=0 | tile_dim0=32
    g1[4] = 128;                                 // tile_dim1=128, tile_dim2=0
    g1[5] = 256;                                 // tensor_dim0_stride=256
    g1[6] = 0;                                   // stride hi / dim1_stride lo
    g1[7] = 0;                                   // dim1_stride hi
    i32x4 g2; g2[0] = 1; g2[1] = 1; g2[2] = 0; g2[3] = 0;   // dims2/3 benign
    i32x4 g3; g3[0] = 0; g3[1] = (1 << 16); g3[2] = 0; g3[3] = 0;
    i32x8 g4 = {};                               // extra group (6-arg form), unused
    __builtin_amdgcn_tensor_load_to_lds(g0, g1, g2, g3, g4, 0);
  };
#else
  uint4 rbuf[2];
  auto gloadB = [&](int k0) {
#pragma unroll
    for (int p = 0; p < 2; ++p) {
      int g = tid + p * 256;
      int n = g >> 2, kq = g & 3;
      rbuf[p] = *(const uint4*)(Bt + (size_t)(n0 + n) * 256 + k0 + kq * 8);
    }
  };
  auto storeB = [&](int buf) {
#pragma unroll
    for (int p = 0; p < 2; ++p) {
      int g = tid + p * 256;
      int n = g >> 2, kq = g & 3;
      *(uint4*)(&lB[buf][n][kq * 8]) = rbuf[p];
    }
  };
#endif

  // Fragment loads per ISA 7.12.2 (16-bit 16x32): lane<16 -> K 0..7 & 16..23,
  // lane>=16 -> K 8..15 & 24..31; two contiguous 16B LDS loads per fragment.
  auto compute = [&](int buf) {
    bf16x16 af[2], bfr[4];
#pragma unroll
    for (int i = 0; i < 2; ++i) {
      const __bf16* p = &lA[buf][wm * 32 + i * 16 + l15][rsel * 8];
      bf16x8 lo = *(const bf16x8*)(p);
      bf16x8 hi = *(const bf16x8*)(p + 16);
      af[i] = __builtin_shufflevector(lo, hi, 0,1,2,3,4,5,6,7,8,9,10,11,12,13,14,15);
    }
#pragma unroll
    for (int j = 0; j < 4; ++j) {
      const __bf16* p = &lB[buf][wn * 64 + j * 16 + l15][rsel * 8];
      bf16x8 lo = *(const bf16x8*)(p);
      bf16x8 hi = *(const bf16x8*)(p + 16);
      bfr[j] = __builtin_shufflevector(lo, hi, 0,1,2,3,4,5,6,7,8,9,10,11,12,13,14,15);
    }
#pragma unroll
    for (int i = 0; i < 2; ++i)
#pragma unroll
      for (int j = 0; j < 4; ++j)
        acc[i][j] = __builtin_amdgcn_wmma_f32_16x16x32_bf16(
            false, af[i], false, bfr[j], (short)0, acc[i][j], false, false);
  };

  // ---- software pipeline: stage(k+1) || wmma(k) ----
  gloadA(0);
#if USE_TDM
  if (wave == 0) tdmB(0, 0);
#else
  gloadB(0);
#endif
  storeA(0);
#if USE_TDM
  if (wave == 0) __builtin_amdgcn_s_wait_tensorcnt(0);
#else
  storeB(0);
#endif
  __syncthreads();

#pragma unroll
  for (int kc = 0; kc < 8; ++kc) {
    const int cur = kc & 1;
    if (kc < 7) {
      gloadA((kc + 1) * 32);
#if USE_TDM
      if (wave == 0) tdmB(cur ^ 1, (kc + 1) * 32);
#else
      gloadB((kc + 1) * 32);
#endif
    }
    compute(cur);
    if (kc < 7) {
      storeA(cur ^ 1);
#if USE_TDM
      if (wave == 0) __builtin_amdgcn_s_wait_tensorcnt(0);
#else
      storeB(cur ^ 1);
#endif
    }
    __syncthreads();
  }

  // Store per C/D layout: vec element e -> row base + rsel*8 + e, col = l15.
#pragma unroll
  for (int i = 0; i < 2; ++i) {
    int rbase = m0 + wm * 32 + i * 16 + rsel * 8;
#pragma unroll
    for (int j = 0; j < 4; ++j) {
      int col = n0 + wn * 64 + j * 16 + l15;
#pragma unroll
      for (int e = 0; e < 8; ++e) {
        int r = rbase + e;
        if (r < M) C[(size_t)r * 256 + col] = acc[i][j][e];
      }
    }
  }
}

// ---------------------------------------------------------------------------
// Per-node attention coefficients: wave per node, shuffle reduction (wave32).
// ---------------------------------------------------------------------------
__global__ __launch_bounds__(256)
void attn_coef(const float* __restrict__ h, const float* __restrict__ a_s,
               const float* __restrict__ a_d, float* __restrict__ osrc,
               float* __restrict__ odst, int N) {
  int wave = threadIdx.x >> 5, lane = threadIdx.x & 31;
  int node = blockIdx.x * 8 + wave;
  if (node >= N) return;
  const float4* hp = (const float4*)(h + (size_t)node * 256);
  const float4* sp = (const float4*)a_s;
  const float4* dp = (const float4*)a_d;
  float s = 0.f, d = 0.f;
#pragma unroll
  for (int q = 0; q < 2; ++q) {
    int v = lane * 2 + q;
    float4 hv = hp[v], sv = sp[v], dv = dp[v];
    s += hv.x * sv.x + hv.y * sv.y + hv.z * sv.z + hv.w * sv.w;
    d += hv.x * dv.x + hv.y * dv.y + hv.z * dv.z + hv.w * dv.w;
  }
#pragma unroll
  for (int off = 16; off; off >>= 1) {
    s += __shfl_down(s, off, 32);
    d += __shfl_down(d, off, 32);
  }
  if (lane == 0) { osrc[node] = s; odst[node] = d; }
}

__global__ void zero_f32(float* p, size_t n) {
  size_t i = (size_t)blockIdx.x * 256 + threadIdx.x;
  if (i < n) p[i] = 0.f;
}

__global__ void init_md(unsigned int* m, float* den, int n) {
  int i = blockIdx.x * 256 + threadIdx.x;
  if (i < n) { m[i] = FENC_NEG_INF; den[i] = 0.f; }
}

// Pass 1: e = leaky_relu(a_src[src]+a_dst[dst]); segment-max into m[dst].
__global__ void epass1(const int* __restrict__ src, const int* __restrict__ dst,
                       const float* __restrict__ as_, const float* __restrict__ ad_,
                       float* __restrict__ ebuf, unsigned int* __restrict__ m, int E) {
  int i = blockIdx.x * 256 + threadIdx.x;
  if (i >= E) return;
  float e = as_[src[i]] + ad_[dst[i]];
  e = (e >= 0.f) ? e : 0.2f * e;
  ebuf[i] = e;
  atomicMax(m + dst[i], fenc(e));
}

// Pass 2: ex = exp(e - m[dst]); segment-sum into den[dst]; ebuf := ex.
__global__ void epass2(const int* __restrict__ dst, float* __restrict__ ebuf,
                       const unsigned int* __restrict__ m, float* __restrict__ den, int E) {
  int i = blockIdx.x * 256 + threadIdx.x;
  if (i >= E) return;
  int d = dst[i];
  float ex = __expf(ebuf[i] - fdec(m[d]));
  ebuf[i] = ex;
  atomicAdd(den + d, ex);
}

// Pass 3: acc[dst,:] += (ex/den[dst]) * h[src,:]. One wave per edge, 8 f32/lane.
__global__ __launch_bounds__(256)
void epass3(const int* __restrict__ src, const int* __restrict__ dst,
            const float* __restrict__ exb, const float* __restrict__ den,
            const float* __restrict__ h, float* __restrict__ acc, int E) {
  int wid  = (int)((blockIdx.x * 256 + threadIdx.x) >> 5);
  int lane = threadIdx.x & 31;
  if (wid >= E) return;
  int s = src[wid], d = dst[wid];
  float alpha = exb[wid] / den[d];
  const float4* hp = (const float4*)(h + (size_t)s * 256);
  float* ap = acc + (size_t)d * 256 + lane * 8;
  float4 v0 = hp[lane * 2], v1 = hp[lane * 2 + 1];
  atomicAdd(ap + 0, alpha * v0.x);
  atomicAdd(ap + 1, alpha * v0.y);
  atomicAdd(ap + 2, alpha * v0.z);
  atomicAdd(ap + 3, alpha * v0.w);
  atomicAdd(ap + 4, alpha * v1.x);
  atomicAdd(ap + 5, alpha * v1.y);
  atomicAdd(ap + 6, alpha * v1.z);
  atomicAdd(ap + 7, alpha * v1.w);
}

// acc = relu(acc + sum_t b[t,:])
__global__ void bias_relu(float* __restrict__ acc, const float* __restrict__ b, size_t total) {
  size_t i = (size_t)blockIdx.x * 256 + threadIdx.x;
  if (i >= total) return;
  int c = (int)(i & 255);
  float v = acc[i] + b[c] + b[256 + c] + b[512 + c];
  acc[i] = v > 0.f ? v : 0.f;
}

// out[n] = sigmoid(h2[n,:].Wh + bh)
__global__ __launch_bounds__(256)
void head(const float* __restrict__ h2, const float* __restrict__ Wh,
          const float* __restrict__ bh, float* __restrict__ out, int N) {
  int wave = threadIdx.x >> 5, lane = threadIdx.x & 31;
  int node = blockIdx.x * 8 + wave;
  if (node >= N) return;
  const float4* hp = (const float4*)(h2 + (size_t)node * 256);
  const float4* wp = (const float4*)Wh;
  float s = 0.f;
#pragma unroll
  for (int q = 0; q < 2; ++q) {
    float4 hv = hp[lane * 2 + q], wv = wp[lane * 2 + q];
    s += hv.x * wv.x + hv.y * wv.y + hv.z * wv.z + hv.w * wv.w;
  }
#pragma unroll
  for (int off = 16; off; off >>= 1) s += __shfl_down(s, off, 32);
  if (lane == 0) out[node] = 1.f / (1.f + __expf(-(s + bh[0])));
}

// ---------------------------------------------------------------------------
static void run_layer(const float* X, const int* ei, const float* W,
                      const float* as_, const float* ad_, const float* b,
                      float* h, float* acc, float* a_s, float* a_d,
                      unsigned int* m, float* den, float* ebuf, __bf16* wt,
                      const float* zpad, int N, int E, hipStream_t stream) {
  const size_t NH = (size_t)N * 256;
  zero_f32<<<dim3((unsigned)((NH + 255) / 256)), 256, 0, stream>>>(acc, NH);
  dim3 ggrid((N + 127) / 128, 2);
  dim3 tgrid(8, 8);
  for (int t = 0; t < 3; ++t) {
    conv_transpose_w<<<tgrid, 256, 0, stream>>>(W + (size_t)t * 256 * 256, wt);
    wmma_gemm_256<<<ggrid, 256, 0, stream>>>(X, wt, h, zpad, N);
    attn_coef<<<(N + 7) / 8, 256, 0, stream>>>(h, as_ + t * 256, ad_ + t * 256, a_s, a_d, N);
    init_md<<<(N + 255) / 256, 256, 0, stream>>>(m, den, N);
    const int* src = ei + (size_t)t * 2 * E;
    const int* dst = src + E;
    epass1<<<(E + 255) / 256, 256, 0, stream>>>(src, dst, a_s, a_d, ebuf, m, E);
    epass2<<<(E + 255) / 256, 256, 0, stream>>>(dst, ebuf, m, den, E);
    epass3<<<(E + 7) / 8, 256, 0, stream>>>(src, dst, ebuf, den, h, acc, E);
  }
  bias_relu<<<(unsigned)((NH + 255) / 256), 256, 0, stream>>>(acc, b, NH);
}

extern "C" void kernel_launch(void* const* d_in, const int* in_sizes, int n_in,
                              void* d_out, int out_size, void* d_ws, size_t ws_size,
                              hipStream_t stream) {
  const float* z   = (const float*)d_in[0];
  const int*   ei  = (const int*)d_in[1];
  const float* W1  = (const float*)d_in[2];
  const float* as1 = (const float*)d_in[3];
  const float* ad1 = (const float*)d_in[4];
  const float* b1  = (const float*)d_in[5];
  const float* W2  = (const float*)d_in[6];
  const float* as2 = (const float*)d_in[7];
  const float* ad2 = (const float*)d_in[8];
  const float* b2  = (const float*)d_in[9];
  const float* Wh  = (const float*)d_in[10];
  const float* bh  = (const float*)d_in[11];
  float* out = (float*)d_out;

  const int N = in_sizes[0] / 256;     // 50000
  const int E = in_sizes[1] / 6;       // 250000 (T=3, src+dst)
  const size_t NH = (size_t)N * 256;

  // Workspace layout (~157 MB).
  float* h    = (float*)d_ws;          // [N,256] per-type dense output
  float* acc1 = h + NH;                // [N,256] layer-1 accumulator / layer-2 input
  float* acc2 = acc1 + NH;             // [N,256] layer-2 accumulator
  float* a_s  = acc2 + NH;             // [N]
  float* a_d  = a_s + N;               // [N]
  unsigned int* m = (unsigned int*)(a_d + N);  // [N] encoded segment max
  float* den  = (float*)(m + N);       // [N]
  float* ebuf = den + N;               // [E] e then ex (reused)
  __bf16* wt  = (__bf16*)(ebuf + E);   // [256,256] bf16 transposed W
  float* zpad = (float*)(wt + 65536);  // [256] zero page for OOB GEMM rows

  zero_f32<<<1, 256, 0, stream>>>(zpad, 256);
  run_layer(z,    ei, W1, as1, ad1, b1, h, acc1, a_s, a_d, m, den, ebuf, wt, zpad, N, E, stream);
  run_layer(acc1, ei, W2, as2, ad2, b2, h, acc2, a_s, a_d, m, den, ebuf, wt, zpad, N, E, stream);
  head<<<(N + 7) / 8, 256, 0, stream>>>(acc2, Wh, bh, out, N);
}
